// Attention_16183436771634
// MI455X (gfx1250) — matrix operand order repaired and checked by reading, not compiled, not run on hardware
//
#include <hip/hip_runtime.h>

// ---------------------------------------------------------------------------
// Attention layer for MI455X (gfx1250, wave32, WMMA).
// B=1, L=2048, DIM=4096, H=32, KV=8, HD=128, REP=4, causal.
// Outputs (concatenated): out@wo [2048*4096], roped-repeated K [32*2048*128],
// repeated V [32*2048*128].
// Matmuls: v_wmma_f32_16x16x32_bf16 (f32 -> bf16 at tile edge, f32 accum).
// GEMM: 64x64 per wave (16 WMMAs / K-step), tiles staged in LDS via
// global_load_async_to_lds_b128 when the toolchain exposes it.
// ---------------------------------------------------------------------------

typedef __bf16 bf16_t;
typedef __attribute__((ext_vector_type(16))) __bf16 v16bf;
typedef __attribute__((ext_vector_type(8)))  float  v8f;
typedef __attribute__((ext_vector_type(4)))  int    v4i_t;

#define LQ    2048
#define DIMX  4096
#define NH    32
#define NKV   8
#define HDIM  128
#define REPF  4
#define LOG10000 9.210340371976184f
#define QSCALE 0.08838834764831845f   // 128^-0.5

#if defined(__has_builtin)
#if __has_builtin(__builtin_amdgcn_global_load_async_to_lds_b128) && \
    __has_builtin(__builtin_amdgcn_s_wait_asynccnt)
#define USE_ASYNC_LDS 1
#endif
#endif

#ifdef USE_ASYNC_LDS
typedef __attribute__((address_space(1))) v4i_t g_v4i;
typedef __attribute__((address_space(3))) v4i_t l_v4i;
#endif

// Copy 16 bytes global -> LDS (async when available).
__device__ __forceinline__ void stage16(const float* g, float* l) {
#ifdef USE_ASYNC_LDS
    __builtin_amdgcn_global_load_async_to_lds_b128(
        (g_v4i*)(unsigned long long)g, (l_v4i*)l, 0, 0);
#else
    *(float4*)l = *(const float4*)g;
#endif
}

__device__ __forceinline__ void stage_wait() {
#ifdef USE_ASYNC_LDS
    __builtin_amdgcn_s_wait_asynccnt(0);
#endif
}

// A-fragment (16-bit, 16x32) K index for element e (0..15), lane-half hf.
__device__ __forceinline__ int amap(int e, int hf) {
    return (e < 8) ? (hf * 8 + e) : (16 + hf * 8 + (e - 8));
}
// B-fragment (16-bit, 32x16) K index: same K order as amap.
__device__ __forceinline__ int bmap(int e, int hf) {
    return (e < 8) ? (hf * 8 + e) : (16 + hf * 8 + (e - 8));
}

__device__ __forceinline__ v8f wmma_bf16(v16bf a, v16bf b, v8f c) {
    return __builtin_amdgcn_wmma_f32_16x16x32_bf16(
        /*neg_a=*/false, a, /*neg_b=*/false, b,
        /*c_mod=*/(short)0, c, /*reuse_a=*/false, /*reuse_b=*/false);
}

// ---------------------------------------------------------------------------
// Generic GEMM: C[M,N] = A[M,K] * B[K,N], f32 in/out, bf16 WMMA inside.
// Block = 256 threads = 8 waves (2 x 4 grid of 64x64 wave tiles).
// Block tile = 128 x 256, K stepped by 32, A/B tiles staged in LDS (48 KB).
// Each wave: 4x4 WMMA tiles -> 16 WMMAs per K-step, 4x fragment reuse.
// ---------------------------------------------------------------------------
__global__ __launch_bounds__(256)
void gemm_bf16_f32(const float* __restrict__ A, const float* __restrict__ B,
                   float* __restrict__ C, int M, int N, int K) {
    __shared__ float Atile[128 * 32];   // 16 KB
    __shared__ float Btile[32 * 256];   // 32 KB

    const int t    = threadIdx.x;
    const int lane = t & 31;
    const int wave = t >> 5;
    const int lo   = lane & 15;
    const int hf   = lane >> 4;
    const int wrow = (wave >> 2) * 64;               // wave row offset in tile
    const int wcol = (wave & 3) * 64;                // wave col offset in tile
    const int rowB = blockIdx.y * 128;               // block row base
    const int colB = blockIdx.x * 256;               // block col base

    // Staging assignments (256 threads):
    // A: 128x32 f32 -> thread copies 16 floats (row t>>1, cols (t&1)*16..)
    // B: 32x256 f32 -> thread copies 32 floats (row t>>3, cols (t&7)*32..)
    const int ar = t >> 1, ac = (t & 1) * 16;
    const int br = t >> 3, bc = (t & 7) * 32;

    const v8f vzero = {0.f, 0.f, 0.f, 0.f, 0.f, 0.f, 0.f, 0.f};
    v8f acc[4][4];
#pragma unroll
    for (int tr = 0; tr < 4; ++tr)
#pragma unroll
        for (int tc = 0; tc < 4; ++tc) acc[tr][tc] = vzero;

    for (int kk = 0; kk < K; kk += 32) {
        // ---- stage A and B tiles into LDS
        const float* ag = A + (size_t)(rowB + ar) * K + kk + ac;
#pragma unroll
        for (int u = 0; u < 4; ++u)
            stage16(ag + 4 * u, &Atile[ar * 32 + ac + 4 * u]);
        const float* bg = B + (size_t)(kk + br) * N + colB + bc;
#pragma unroll
        for (int u = 0; u < 8; ++u)
            stage16(bg + 4 * u, &Btile[br * 256 + bc + 4 * u]);
        stage_wait();
        __syncthreads();

        // ---- A fragments: contiguous LDS reads (kept live, reused 4x)
        v16bf afrag[4];
#pragma unroll
        for (int tr = 0; tr < 4; ++tr) {
            const float* ap = &Atile[(wrow + tr * 16 + lo) * 32];
#pragma unroll
            for (int e = 0; e < 16; ++e)
                afrag[tr][e] = (__bf16)ap[amap(e, hf)];
        }
        // ---- B fragments: one at a time, each feeds 4 WMMAs
#pragma unroll
        for (int tc = 0; tc < 4; ++tc) {
            const int n = wcol + tc * 16 + lo;
            v16bf bfrag;
#pragma unroll
            for (int e = 0; e < 16; ++e)
                bfrag[e] = (__bf16)Btile[bmap(e, hf) * 256 + n];
#pragma unroll
            for (int tr = 0; tr < 4; ++tr)
                acc[tr][tc] = wmma_bf16(afrag[tr], bfrag, acc[tr][tc]);
        }
        __syncthreads();
    }

#pragma unroll
    for (int tr = 0; tr < 4; ++tr)
#pragma unroll
        for (int tc = 0; tc < 4; ++tc)
#pragma unroll
            for (int r = 0; r < 8; ++r) {
                const int row = rowB + wrow + tr * 16 + r + 8 * hf;
                const int col = colB + wcol + tc * 16 + lo;
                __builtin_nontemporal_store(acc[tr][tc][r],
                                            &C[(size_t)row * N + col]);
            }
}

// ---------------------------------------------------------------------------
// RoPE for Q: rope + fold in softmax scale, write bf16 (L, H*HD) row-major.
// grid = (NH, LQ), block = 64 (one thread per even/odd pair).
// ---------------------------------------------------------------------------
__global__ __launch_bounds__(64)
void rope_q_kernel(const float* __restrict__ qf, bf16_t* __restrict__ qbf) {
    const int h = blockIdx.x, l = blockIdx.y, i = threadIdx.x;
    const float inv_freq = __expf(-(2.0f * i) * (LOG10000 / (float)HDIM));
    const float ang = (float)l * inv_freq;
    float s, c;
    __sincosf(ang, &s, &c);
    const size_t base = (size_t)l * DIMX + (size_t)h * HDIM;
    const float x1 = qf[base + 2 * i], x2 = qf[base + 2 * i + 1];
    qbf[base + 2 * i]     = (__bf16)((x1 * c - x2 * s) * QSCALE);
    qbf[base + 2 * i + 1] = (__bf16)((x1 * s + x2 * c) * QSCALE);
}

// ---------------------------------------------------------------------------
// RoPE K + expand K/V outputs:
//  - kbf: roped K, bf16, (L, KV*HD) row-major (attention B operand)
//  - vTbf: V transposed, bf16, (KV, HD, L)   (attention PV B operand)
//  - kout/vout: f32, (H, L, HD), KV heads repeated REP times (d_out regions)
// grid = (NKV, LQ), block = 64.
// ---------------------------------------------------------------------------
__global__ __launch_bounds__(64)
void rope_kv_kernel(const float* __restrict__ kf, const float* __restrict__ vf,
                    bf16_t* __restrict__ kbf, bf16_t* __restrict__ vTbf,
                    float* __restrict__ kout, float* __restrict__ vout) {
    const int kvh = blockIdx.x, l = blockIdx.y, i = threadIdx.x;
    const size_t base = (size_t)l * (NKV * HDIM) + (size_t)kvh * HDIM;

    const float inv_freq = __expf(-(2.0f * i) * (LOG10000 / (float)HDIM));
    const float ang = (float)l * inv_freq;
    float s, c;
    __sincosf(ang, &s, &c);
    const float x1 = kf[base + 2 * i], x2 = kf[base + 2 * i + 1];
    const float r1 = x1 * c - x2 * s;
    const float r2 = x1 * s + x2 * c;
    kbf[base + 2 * i]     = (__bf16)r1;
    kbf[base + 2 * i + 1] = (__bf16)r2;
#pragma unroll
    for (int r = 0; r < REPF; ++r) {
        const int h = kvh * REPF + r;
        const size_t ob = ((size_t)h * LQ + l) * HDIM;
        __builtin_nontemporal_store(r1, &kout[ob + 2 * i]);
        __builtin_nontemporal_store(r2, &kout[ob + 2 * i + 1]);
    }

#pragma unroll
    for (int tt = 0; tt < 2; ++tt) {
        const int j = i + tt * 64;
        const float v = vf[base + j];
        vTbf[((size_t)kvh * HDIM + j) * LQ + l] = (__bf16)v;
#pragma unroll
        for (int r = 0; r < REPF; ++r) {
            const int h = kvh * REPF + r;
            __builtin_nontemporal_store(v, &vout[((size_t)h * LQ + l) * HDIM + j]);
        }
    }
}

// ---------------------------------------------------------------------------
// Flash-style causal attention. One wave per (head, 16-query tile).
// Scores: QK^T via 8 WMMAs per 32-key block; online softmax with half-wave
// shuffle reductions; probs re-laid-out through LDS into an A fragment;
// PV via 8 WMMAs per key block into the 16x128 f32 accumulator.
// ---------------------------------------------------------------------------
__global__ __launch_bounds__(32)
void attn_kernel(const bf16_t* __restrict__ qbf, const bf16_t* __restrict__ kbf,
                 const bf16_t* __restrict__ vTbf, float* __restrict__ ao) {
    __shared__ bf16_t P[16 * 32];

    const int h = blockIdx.y;
    const int qbase = blockIdx.x * 16;
    const int kvh = h / REPF;
    const int lane = threadIdx.x & 31;
    const int lo = lane & 15;
    const int hf = lane >> 4;

    v16bf qfrag[4];
#pragma unroll
    for (int f = 0; f < 4; ++f) {
        const bf16_t* qp = qbf + (size_t)(qbase + lo) * DIMX
                               + (size_t)h * HDIM + f * 32;
#pragma unroll
        for (int e = 0; e < 16; ++e) qfrag[f][e] = qp[amap(e, hf)];
    }

    const v8f vzero = {0.f, 0.f, 0.f, 0.f, 0.f, 0.f, 0.f, 0.f};
    v8f oacc[8];
#pragma unroll
    for (int nc = 0; nc < 8; ++nc) oacc[nc] = vzero;
    float mrun[8], srun[8];
#pragma unroll
    for (int r = 0; r < 8; ++r) { mrun[r] = -3.0e38f; srun[r] = 0.f; }

    const int kend = qbase + 16;            // causal: keys < kend
    for (int jb = 0; jb < kend; jb += 32) {
        v8f sc0 = vzero, sc1 = vzero;
#pragma unroll
        for (int f = 0; f < 4; ++f) {
            v16bf b0, b1;
            const bf16_t* kp0 = kbf + (size_t)(jb + lo) * (NKV * HDIM)
                                    + (size_t)kvh * HDIM + f * 32;
            const bf16_t* kp1 = kbf + (size_t)(jb + 16 + lo) * (NKV * HDIM)
                                    + (size_t)kvh * HDIM + f * 32;
#pragma unroll
            for (int e = 0; e < 16; ++e) {
                b0[e] = kp0[bmap(e, hf)];
                b1[e] = kp1[bmap(e, hf)];
            }
            sc0 = wmma_bf16(qfrag[f], b0, sc0);
            sc1 = wmma_bf16(qfrag[f], b1, sc1);
        }

        if (jb + 31 > qbase) {
#pragma unroll
            for (int r = 0; r < 8; ++r) {
                const int q = qbase + r + 8 * hf;
                if (jb + lo > q)      sc0[r] += -1.0e9f;
                if (jb + 16 + lo > q) sc1[r] += -1.0e9f;
            }
        }

#pragma unroll
        for (int r = 0; r < 8; ++r) {
            float tmax = fmaxf(sc0[r], sc1[r]);
#pragma unroll
            for (int off = 8; off >= 1; off >>= 1)
                tmax = fmaxf(tmax, __shfl_xor(tmax, off, 32));
            const float nm = fmaxf(mrun[r], tmax);
            const float corr = __expf(mrun[r] - nm);
            const float p0 = __expf(sc0[r] - nm);
            const float p1 = __expf(sc1[r] - nm);
            float rs = p0 + p1;
#pragma unroll
            for (int off = 8; off >= 1; off >>= 1)
                rs += __shfl_xor(rs, off, 32);
            srun[r] = srun[r] * corr + rs;
            mrun[r] = nm;
#pragma unroll
            for (int nc = 0; nc < 8; ++nc) oacc[nc][r] *= corr;
            P[(r + 8 * hf) * 32 + lo]      = (__bf16)p0;
            P[(r + 8 * hf) * 32 + 16 + lo] = (__bf16)p1;
        }
        __syncthreads();

        v16bf pfrag;
#pragma unroll
        for (int e = 0; e < 16; ++e) pfrag[e] = P[lo * 32 + amap(e, hf)];

#pragma unroll
        for (int nc = 0; nc < 8; ++nc) {
            v16bf vfrag;
            const bf16_t* vp = vTbf + ((size_t)kvh * HDIM + nc * 16 + lo) * LQ + jb;
#pragma unroll
            for (int e = 0; e < 16; ++e) vfrag[e] = vp[bmap(e, hf)];
            oacc[nc] = wmma_bf16(pfrag, vfrag, oacc[nc]);
        }
        __syncthreads();
    }

#pragma unroll
    for (int nc = 0; nc < 8; ++nc)
#pragma unroll
        for (int r = 0; r < 8; ++r) {
            const int row = qbase + r + 8 * hf;
            const int col = h * HDIM + nc * 16 + lo;
            __builtin_nontemporal_store(oacc[nc][r] / srun[r],
                                        &ao[(size_t)row * DIMX + col]);
        }
}

// ---------------------------------------------------------------------------
extern "C" void kernel_launch(void* const* d_in, const int* in_sizes, int n_in,
                              void* d_out, int out_size, void* d_ws, size_t ws_size,
                              hipStream_t stream) {
    const float* x  = (const float*)d_in[0];
    const float* wq = (const float*)d_in[1];
    const float* wk = (const float*)d_in[2];
    const float* wv = (const float*)d_in[3];
    const float* wo = (const float*)d_in[4];
    // d_in[5] (mask) unused: causal mask computed inline.

    char* ws = (char*)d_ws;
    float*  q_ws  = (float*)(ws);                        // 32 MB
    float*  k_ws  = (float*)(ws + (size_t)(32u << 20));  //  8 MB
    float*  v_ws  = (float*)(ws + (size_t)(40u << 20));  //  8 MB
    float*  ao_ws = (float*)(ws + (size_t)(48u << 20));  // 32 MB
    bf16_t* q_bf  = (bf16_t*)(ws + (size_t)(80u << 20)); // 16 MB
    bf16_t* k_bf  = (bf16_t*)(ws + (size_t)(96u << 20)); //  4 MB
    bf16_t* vT_bf = (bf16_t*)(ws + (size_t)(100u << 20));//  4 MB (104 MB total)

    float* outp = (float*)d_out;
    float* kout = outp + (size_t)LQ * DIMX;
    float* vout = kout + (size_t)NH * LQ * HDIM;

    const dim3 blk(256);
    gemm_bf16_f32<<<dim3(DIMX / 256, LQ / 128), blk, 0, stream>>>(
        x, wq, q_ws, LQ, DIMX, DIMX);
    gemm_bf16_f32<<<dim3((NKV * HDIM) / 256, LQ / 128), blk, 0, stream>>>(
        x, wk, k_ws, LQ, NKV * HDIM, DIMX);
    gemm_bf16_f32<<<dim3((NKV * HDIM) / 256, LQ / 128), blk, 0, stream>>>(
        x, wv, v_ws, LQ, NKV * HDIM, DIMX);

    rope_q_kernel<<<dim3(NH, LQ), 64, 0, stream>>>(q_ws, q_bf);
    rope_kv_kernel<<<dim3(NKV, LQ), 64, 0, stream>>>(
        k_ws, v_ws, k_bf, vT_bf, kout, vout);

    attn_kernel<<<dim3(LQ / 16, NH), 32, 0, stream>>>(q_bf, k_bf, vT_bf, ao_ws);

    gemm_bf16_f32<<<dim3(DIMX / 256, LQ / 128), blk, 0, stream>>>(
        ao_ws, wo, outp, LQ, DIMX, DIMX);
}
